// NodeUpdateNetwork_506806141134
// MI455X (gfx1250) — matrix operand order repaired and checked by reading, not compiled, not run on hardware
//
#include <hip/hip_runtime.h>

// MI455X / gfx1250, wave32. bf16 WMMA 16x16x32, f32 accumulate.
//
// Roofline: edge_feat = 537 MB must stream once from HBM (23.3 TB/s -> ~23us
// floor). FLOPs ~18 GF -> memory-bound. Structure: single pass over edge_feat,
// diagonal mask + |.| row-sum computed in-registers during the stream, L1
// normalization deferred to a per-row output scale (normalization commutes
// with the matmul). K-loop is register double-buffered so WMMA + cvt VALU of
// chunk i co-execute with the 12 in-flight global loads of chunk i+1.

typedef __attribute__((ext_vector_type(16))) __bf16          v16bf;
typedef __attribute__((ext_vector_type(8)))  float           v8f;
typedef __attribute__((ext_vector_type(8)))  unsigned int    v8u;

#define T_DIM 4
#define N_DIM 4096
#define D_DIM 64
#define E_DIM 2
#define H0_DIM 128
#define CIN_DIM 192
#define OUT_DIM 64

__device__ __forceinline__ unsigned short f2bf(float f) {
    return __builtin_bit_cast(unsigned short, (__bf16)f);
}
__device__ __forceinline__ float lrelu(float x) { return x > 0.f ? x : 0.01f * x; }

// 16 f32 -> bf16 A-fragment, preferring the packed hardware convert.
__device__ __forceinline__ v16bf cvt16(const float* v) {
    v16bf a;
#if __has_builtin(__builtin_amdgcn_cvt_pk_bf16_f32)
#pragma unroll
    for (int i = 0; i < 8; ++i) {
        auto p = __builtin_amdgcn_cvt_pk_bf16_f32(v[2 * i], v[2 * i + 1]);
        a[2 * i]     = p[0];
        a[2 * i + 1] = p[1];
    }
#else
#pragma unroll
    for (int i = 0; i < 16; ++i) a[i] = (__bf16)v[i];
#endif
    return a;
}

// ---------------------------------------------------------------------------
// Prep 1: node_feat [T][N][64] f32 -> node_T [T][64][N] bf16 (WMMA B-frags
// become 16 contiguous K values per lane).
__global__ void k_prep_node(const float* __restrict__ node,
                            unsigned short* __restrict__ nodeT) {
    int id = blockIdx.x * blockDim.x + threadIdx.x;   // T*N*D = 1048576 exact
    int d = id & 63;
    int n = (id >> 6) & (N_DIM - 1);
    int t = id >> 18;
    nodeT[((size_t)(t * 64 + d)) * N_DIM + n] = f2bf(node[id]);
}

// Prep 2: weights f32 -> bf16 (row-major kept: B-frag lane reads one weight row).
__global__ void k_prep_w(const float* __restrict__ w0, const float* __restrict__ w1,
                         unsigned short* __restrict__ w0b, unsigned short* __restrict__ w1b) {
    int id = blockIdx.x * blockDim.x + threadIdx.x;   // 32768 exact
    if (id < H0_DIM * CIN_DIM) w0b[id] = f2bf(w0[id]);
    else {
        int j = id - H0_DIM * CIN_DIM;
        w1b[j] = f2bf(w1[j]);
    }
}

// ---------------------------------------------------------------------------
// Aggregation helpers: one K-chunk = 32 columns of the edge row (one full
// 128B cacheline per lane-pair) + 4 B fragments (d = 0..63).
__device__ __forceinline__ void loadc(const float* __restrict__ arow,
                                      const unsigned short* __restrict__ bbase,
                                      int m, int half, int kbase,
                                      float4* f, v8u* b) {
    const float4* pa = (const float4*)(arow + kbase + half * 16);
    const float4* pb = (const float4*)(arow + kbase + half * 16 + 8);
    f[0] = pa[0]; f[1] = pa[1]; f[2] = pb[0]; f[3] = pb[1];
#pragma unroll
    for (int j = 0; j < 4; ++j)
        b[j] = *(const v8u*)(bbase + (size_t)(j * 16 + m) * N_DIM + kbase + half * 16);
}

__device__ __forceinline__ void proc(int kbase, bool dchunk, int diag, int half,
                                     const float4* f, const v8u* b,
                                     v8f* acc, float* rs) {
    float av[16] = { f[0].x, f[0].y, f[0].z, f[0].w, f[1].x, f[1].y, f[1].z, f[1].w,
                     f[2].x, f[2].y, f[2].z, f[2].w, f[3].x, f[3].y, f[3].z, f[3].w };
    if (dchunk) {            // wave-uniform: only 1 of 128 chunks holds diagonals
#pragma unroll
        for (int i = 0; i < 16; ++i) {
            int k = kbase + half * 16 + i;
            if (k == diag) av[i] = 0.f;
        }
    }
    // 4 independent partial |.|-sums: breaks the serial VALU dependency chain
    // so these adds co-issue with the WMMA/cvt stream.
#pragma unroll
    for (int i = 0; i < 16; ++i) rs[i & 3] += fabsf(av[i]);
    v16bf a = cvt16(av);
#pragma unroll
    for (int j = 0; j < 4; ++j)
        acc[j] = __builtin_amdgcn_wmma_f32_16x16x32_bf16(
            false, a, false, __builtin_bit_cast(v16bf, b[j]),
            (short)0, acc[j], false, false);
}

// One block per (t, e, 16-row tile). 8 waves split K=4096 (16 chunks each).
// C[16x64] = maskdiag(edge rows) @ node ; scaled by 1/L1rowsum at the end.
__launch_bounds__(256)
__global__ void k_aggr(const float* __restrict__ edge,
                       const unsigned short* __restrict__ nodeT,
                       float* __restrict__ aggr) {
    __shared__ float s_partial[8][16][64];   // 32 KB
    __shared__ float s_rowsum[16];

    const int lane = threadIdx.x & 31;
    const int wave = threadIdx.x >> 5;
    const int m    = lane & 15;              // row within tile (A/M, C/N lane)
    const int half = lane >> 4;

    const int tile = blockIdx.x;             // T*E*256 = 2048
    const int rt   = tile & 255;
    const int te   = tile >> 8;
    const int e    = te & 1;
    const int t    = te >> 1;
    const int n0   = rt << 4;

    if (threadIdx.x < 16) s_rowsum[threadIdx.x] = 0.f;
    __syncthreads();

    const float* arow = edge + (((size_t)(t * E_DIM + e) * N_DIM + (n0 + m)) * N_DIM);
    const unsigned short* bbase = nodeT + (size_t)t * 64 * N_DIM;
    const int diag  = n0 + m;
    const int kdiag = n0 >> 5;               // the single chunk holding diagonals

    v8f acc[4];
#pragma unroll
    for (int j = 0; j < 4; ++j)
#pragma unroll
        for (int r = 0; r < 8; ++r) acc[j][r] = 0.f;
    float rs[4] = {0.f, 0.f, 0.f, 0.f};

    // Register double-buffered K loop: 16 chunks/wave, unrolled by 2.
    // Prefetch address clamped (branchless) on the final iteration: the
    // redundant reload keeps the software pipeline fully regular.
    float4 fA[4], fB[4];
    v8u    bA[4], bB[4];
    loadc(arow, bbase, m, half, wave << 5, fA, bA);
    for (int kb = wave; kb < 128; kb += 16) {
        loadc(arow, bbase, m, half, (kb + 8) << 5, fB, bB);
        proc(kb << 5, kb == kdiag, diag, half, fA, bA, acc, rs);
        const int knext = (kb + 16 < 128) ? ((kb + 16) << 5) : (wave << 5);
        loadc(arow, bbase, m, half, knext, fA, bA);
        proc((kb + 8) << 5, (kb + 8) == kdiag, diag, half, fB, bB, acc, rs);
    }

    atomicAdd(&s_rowsum[m], (rs[0] + rs[1]) + (rs[2] + rs[3]));
#pragma unroll
    for (int j = 0; j < 4; ++j)
#pragma unroll
        for (int r = 0; r < 8; ++r)
            s_partial[wave][r + half * 8][j * 16 + m] = acc[j][r];
    __syncthreads();

    // reduce 8 wave-partials, apply deferred 1/denom row scale
    for (int idx = threadIdx.x; idx < 16 * 64; idx += 256) {
        int row = idx >> 6, d = idx & 63;
        float s = 0.f;
#pragma unroll
        for (int w = 0; w < 8; ++w) s += s_partial[w][row][d];
        float denom = s_rowsum[row];
        denom = denom > 1e-12f ? denom : 1e-12f;
        aggr[((size_t)t * N_DIM + n0 + row) * (E_DIM * D_DIM) + e * D_DIM + d] = s / denom;
    }
}

// ---------------------------------------------------------------------------
// MLP: per-wave 16-node tile. x = [node | aggr_e0 | aggr_e1] gathered straight
// from global per K-chunk (64-boundary aligns with 32-wide chunks).
// GEMM1: 16x192 @ 192x128 (+LeakyReLU) -> GEMM2: 16x128 @ 128x64 (+LeakyReLU).
__launch_bounds__(256)
__global__ void k_mlp(const float* __restrict__ node, const float* __restrict__ aggr,
                      const unsigned short* __restrict__ w0b,
                      const unsigned short* __restrict__ w1b,
                      float* __restrict__ out) {
    __shared__ unsigned short s_h[8][16][H0_DIM];   // 32 KB

    const int lane = threadIdx.x & 31;
    const int wave = threadIdx.x >> 5;
    const int m    = lane & 15;
    const int half = lane >> 4;

    const int tileIdx = blockIdx.x * 8 + wave;      // 0..1023
    const int t  = tileIdx >> 8;
    const int n0 = (tileIdx & 255) << 4;
    const int row = n0 + m;

    const float* nrow = node + ((size_t)t * N_DIM + row) * D_DIM;
    const float* grow = aggr + ((size_t)t * N_DIM + row) * (E_DIM * D_DIM);

    v8f acc1[8];
#pragma unroll
    for (int j = 0; j < 8; ++j)
#pragma unroll
        for (int r = 0; r < 8; ++r) acc1[j][r] = 0.f;

#pragma unroll
    for (int kc = 0; kc < 6; ++kc) {
        const int kbase = kc * 32;
        const float* src = (kbase < 64) ? (nrow + kbase) : (grow + (kbase - 64));
        const float4* pa = (const float4*)(src + half * 16);
        const float4* pb = (const float4*)(src + half * 16 + 8);
        float4 f0 = pa[0], f1 = pa[1], f2 = pb[0], f3 = pb[1];
        float av[16] = { f0.x, f0.y, f0.z, f0.w, f1.x, f1.y, f1.z, f1.w,
                         f2.x, f2.y, f2.z, f2.w, f3.x, f3.y, f3.z, f3.w };
        v16bf a = cvt16(av);

#pragma unroll
        for (int j = 0; j < 8; ++j) {
            const v8u* bp = (const v8u*)(w0b + (size_t)(j * 16 + m) * CIN_DIM
                                         + kbase + half * 16);
            v16bf b = __builtin_bit_cast(v16bf, *bp);
            acc1[j] = __builtin_amdgcn_wmma_f32_16x16x32_bf16(
                false, a, false, b, (short)0, acc1[j], false, false);
        }
    }

    // LeakyReLU -> bf16 h-tile in LDS (cross-lane relayout C-frag -> A-frag)
#pragma unroll
    for (int j = 0; j < 8; ++j)
#pragma unroll
        for (int r = 0; r < 8; ++r)
            s_h[wave][r + half * 8][j * 16 + m] = f2bf(lrelu(acc1[j][r]));
    __syncthreads();

    v8f acc2[4];
#pragma unroll
    for (int j = 0; j < 4; ++j)
#pragma unroll
        for (int r = 0; r < 8; ++r) acc2[j][r] = 0.f;

    const unsigned short* hrow = &s_h[wave][m][0];
#pragma unroll
    for (int kc = 0; kc < 4; ++kc) {
        const int kbase = kc * 32;
        uint4 lo = *(const uint4*)(hrow + kbase + half * 16);
        uint4 hi = *(const uint4*)(hrow + kbase + half * 16 + 8);
        v8u au;
        au[0] = lo.x; au[1] = lo.y; au[2] = lo.z; au[3] = lo.w;
        au[4] = hi.x; au[5] = hi.y; au[6] = hi.z; au[7] = hi.w;
        v16bf a = __builtin_bit_cast(v16bf, au);

#pragma unroll
        for (int j = 0; j < 4; ++j) {
            const v8u* bp = (const v8u*)(w1b + (size_t)(j * 16 + m) * H0_DIM
                                         + kbase + half * 16);
            v16bf b = __builtin_bit_cast(v16bf, *bp);
            acc2[j] = __builtin_amdgcn_wmma_f32_16x16x32_bf16(
                false, a, false, b, (short)0, acc2[j], false, false);
        }
    }

#pragma unroll
    for (int j = 0; j < 4; ++j)
#pragma unroll
        for (int r = 0; r < 8; ++r)
            out[((size_t)t * N_DIM + n0 + r + half * 8) * OUT_DIM + j * 16 + m] =
                lrelu(acc2[j][r]);
}

// ---------------------------------------------------------------------------
extern "C" void kernel_launch(void* const* d_in, const int* in_sizes, int n_in,
                              void* d_out, int out_size, void* d_ws, size_t ws_size,
                              hipStream_t stream) {
    const float* node = (const float*)d_in[0];   // [T,N,64]
    const float* edge = (const float*)d_in[1];   // [T,E,N,N]
    const float* w0   = (const float*)d_in[2];   // [128,192]
    const float* w1   = (const float*)d_in[3];   // [64,128]
    float* out = (float*)d_out;                  // [T,N,64]

    char* ws = (char*)d_ws;
    // node_T bf16: 4*64*4096*2 = 2 MB
    unsigned short* nodeT = (unsigned short*)ws;
    size_t off = (size_t)T_DIM * 64 * N_DIM * 2;
    unsigned short* w0b = (unsigned short*)(ws + off);   off += H0_DIM * CIN_DIM * 2;
    unsigned short* w1b = (unsigned short*)(ws + off);   off += OUT_DIM * H0_DIM * 2;
    off = (off + 255) & ~(size_t)255;
    float* aggr = (float*)(ws + off);            // 4*4096*128*4 = 8 MB

    k_prep_node<<<(T_DIM * N_DIM * D_DIM) / 256, 256, 0, stream>>>(node, nodeT);
    k_prep_w<<<(H0_DIM * CIN_DIM + OUT_DIM * H0_DIM) / 256, 256, 0, stream>>>(w0, w1, w0b, w1b);
    k_aggr<<<T_DIM * E_DIM * (N_DIM / 16), 256, 0, stream>>>(edge, nodeT, aggr);
    k_mlp<<<(T_DIM * N_DIM) / 128, 256, 0, stream>>>(node, aggr, w0b, w1b, out);
}
